// MultiHeadSelfAttention_47820165873956
// MI455X (gfx1250) — matrix operand order repaired and checked by reading, not compiled, not run on hardware
//
#include <hip/hip_runtime.h>
#include <math.h>

#define D_MODEL   1024
#define NUM_HEADS 16
#define D_HEAD    64
#define BATCH     4
#define SEQ       2048

typedef __attribute__((ext_vector_type(16))) __bf16 v16bf;
typedef __attribute__((ext_vector_type(8)))  __bf16 v8bf;
typedef __attribute__((ext_vector_type(4)))  __bf16 v4bf;
typedef __attribute__((ext_vector_type(8)))  float  v8f;
typedef __attribute__((ext_vector_type(4)))  unsigned int v4u;
typedef __attribute__((ext_vector_type(8)))  int    v8i;
typedef __attribute__((ext_vector_type(4)))  int    v4i;

// ---------------------------------------------------------------------------
// WMMA fragment helpers (CDNA5 wave32 layouts, CDNA5 ISA §7.12.2)
// A 16x32 bf16: lane {h,m}: K(j,e,h)=16*(j>>2)+8h+2*(j&3)+(e&1)
//   -> per lane: two contiguous 8-element chunks at k_base+8h and k_base+16+8h
// B 32x16 bf16 (from N x K row-major): lane {h,n}: same K order as A
//   -> per lane: two contiguous 8-element chunks at k_base+8h and k_base+16+8h
// ---------------------------------------------------------------------------
static __device__ __forceinline__ v16bf
load_frag_a_b16(const __bf16* __restrict__ base, int ld, int m_base, int k_base) {
    const int lane = threadIdx.x & 31;
    const int hw = lane >> 4, m = lane & 15;
    const __bf16* row = base + (size_t)(m_base + m) * ld + k_base + 8 * hw;
    const v8bf c0 = *(const v8bf*)(row);
    const v8bf c1 = *(const v8bf*)(row + 16);
    v16bf a;
#pragma unroll
    for (int i = 0; i < 8; ++i) { a[i] = c0[i]; a[8 + i] = c1[i]; }
    return a;
}

static __device__ __forceinline__ v16bf
load_frag_b_b16(const __bf16* __restrict__ base, int ld, int n_base, int k_base) {
    const int lane = threadIdx.x & 31;
    const int hw = lane >> 4, n = lane & 15;
    const __bf16* row = base + (size_t)(n_base + n) * ld + k_base + 8 * hw;
    const v8bf c0 = *(const v8bf*)(row);
    const v8bf c1 = *(const v8bf*)(row + 16);
    v16bf b;
#pragma unroll
    for (int i = 0; i < 8; ++i) { b[i] = c0[i]; b[8 + i] = c1[i]; }
    return b;
}

static __device__ __forceinline__ v8f wmma_bf16(v16bf a, v16bf b, v8f c) {
    return __builtin_amdgcn_wmma_f32_16x16x32_bf16(
        false, a, false, b, (short)0, c, false, false);
}

// ---------------------------------------------------------------------------
// Tensor Data Mover: 2D tile (bf16) global -> LDS.  D# packed per
// CDNA5 ISA (count=1, data_size=2B, type=2).
// This toolchain exposes the 6-arg builtin:
//   (v4u g0, v8i g1, v4i g2, v4i g3, v8i g4, i32 cpol)
// ---------------------------------------------------------------------------
static __device__ __forceinline__ void tdm_load_2d_b16(
    unsigned int lds_addr, const __bf16* gptr,
    unsigned int tensor_d0, unsigned int tensor_d1,
    unsigned int tile_d0, unsigned int tile_d1, unsigned int d0_stride) {
    const unsigned long long ga = (unsigned long long)(uintptr_t)gptr;
    v4u g0;
    g0[0] = 1u;                                        // count=1, no gather
    g0[1] = lds_addr;                                  // lds_addr[31:0]
    g0[2] = (unsigned int)(ga & 0xFFFFFFFFu);          // global_addr[31:0]
    g0[3] = (unsigned int)((ga >> 32) & 0x1FFFFFFu)    // global_addr[56:32]
          | (2u << 30);                                // type=2 (image)
    v8i g1;
    g1[0] = (int)(1u << 16);                           // data_size=1 -> 2 bytes
    g1[1] = (int)((tensor_d0 & 0xFFFFu) << 16);        // tensor_dim0[15:0]
    g1[2] = (int)((tensor_d0 >> 16) | ((tensor_d1 & 0xFFFFu) << 16));
    g1[3] = (int)((tensor_d1 >> 16) | (tile_d0 << 16));// tile_dim0
    g1[4] = (int)(tile_d1 & 0xFFFFu);                  // tile_dim1, tile_dim2=0
    g1[5] = (int)d0_stride;                            // dim0_stride[31:0]
    g1[6] = 0;                                         // stride hi / dim1_stride
    g1[7] = 0;
    const v4i z4 = {0, 0, 0, 0};
    const v8i z8 = {0, 0, 0, 0, 0, 0, 0, 0};
    __builtin_amdgcn_tensor_load_to_lds(g0, g1, z4, z4, z8, 0);
}

// ---------------------------------------------------------------------------
// Kernel 0: f32 -> bf16 conversion (vectorized)
// ---------------------------------------------------------------------------
__global__ __launch_bounds__(256) void cvt_kernel(
    const float* __restrict__ in, __bf16* __restrict__ out, int n4) {
    const int i = blockIdx.x * blockDim.x + threadIdx.x;
    if (i < n4) {
        const float4 f = ((const float4*)in)[i];
        v4bf o;
        o[0] = (__bf16)f.x; o[1] = (__bf16)f.y;
        o[2] = (__bf16)f.z; o[3] = (__bf16)f.w;
        ((v4bf*)out)[i] = o;
    }
}

// ---------------------------------------------------------------------------
// Kernel 1: QKV projections (bf16 in/out).  Wave = 16(M) x 64(N) tile,
// K=1024 in 32 steps of 32, double-buffered fragment loads.
//   Q,K -> [B,H,S,Dh]   V -> transposed [B,H,Dh,S]
// ---------------------------------------------------------------------------
__global__ __launch_bounds__(128) void qkv_proj_kernel(
    const __bf16* __restrict__ xb,
    const __bf16* __restrict__ wq, const __bf16* __restrict__ wk,
    const __bf16* __restrict__ wv,
    __bf16* __restrict__ Q, __bf16* __restrict__ K, __bf16* __restrict__ Vt) {
    const int w   = threadIdx.x >> 5;
    const int wid = blockIdx.x * 4 + w;          // 24576 wave-tiles
    const int mat = wid / 8192;                  // 0=Q 1=K 2=V
    const int rem = wid % 8192;
    const int mt  = rem & 511;
    const int nt  = rem >> 9;
    const __bf16* W = (mat == 0) ? wq : (mat == 1) ? wk : wv;
    const int n0 = nt * 64;

    const __bf16* arow = xb + (size_t)(mt * 16 + (threadIdx.x & 15)) * D_MODEL;

    v8f acc0 = {}, acc1 = {}, acc2 = {}, acc3 = {};
    v16bf a_c  = load_frag_a_b16(xb, D_MODEL, mt * 16, 0);
    v16bf b_c0 = load_frag_b_b16(W, D_MODEL, n0 +  0, 0);
    v16bf b_c1 = load_frag_b_b16(W, D_MODEL, n0 + 16, 0);
    v16bf b_c2 = load_frag_b_b16(W, D_MODEL, n0 + 32, 0);
    v16bf b_c3 = load_frag_b_b16(W, D_MODEL, n0 + 48, 0);

#pragma unroll 2
    for (int ks = 0; ks < D_MODEL / 32 - 1; ++ks) {
        const int kn = (ks + 1) * 32;
        const v16bf a_n  = load_frag_a_b16(xb, D_MODEL, mt * 16, kn);
        const v16bf b_n0 = load_frag_b_b16(W, D_MODEL, n0 +  0, kn);
        const v16bf b_n1 = load_frag_b_b16(W, D_MODEL, n0 + 16, kn);
        const v16bf b_n2 = load_frag_b_b16(W, D_MODEL, n0 + 32, kn);
        const v16bf b_n3 = load_frag_b_b16(W, D_MODEL, n0 + 48, kn);
        __builtin_prefetch(arow + (ks + 2) * 32, 0, 3);
        acc0 = wmma_bf16(a_c, b_c0, acc0);
        acc1 = wmma_bf16(a_c, b_c1, acc1);
        acc2 = wmma_bf16(a_c, b_c2, acc2);
        acc3 = wmma_bf16(a_c, b_c3, acc3);
        a_c = a_n; b_c0 = b_n0; b_c1 = b_n1; b_c2 = b_n2; b_c3 = b_n3;
    }
    acc0 = wmma_bf16(a_c, b_c0, acc0);
    acc1 = wmma_bf16(a_c, b_c1, acc1);
    acc2 = wmma_bf16(a_c, b_c2, acc2);
    acc3 = wmma_bf16(a_c, b_c3, acc3);

    const int lane = threadIdx.x & 31;
    const int hw = lane >> 4, n = lane & 15;
    v8f accs[4] = {acc0, acc1, acc2, acc3};
#pragma unroll
    for (int j = 0; j < 4; ++j) {
#pragma unroll
        for (int r = 0; r < 8; ++r) {
            const int mrow = mt * 16 + r + 8 * hw;
            const int col  = n0 + j * 16 + n;
            const int b  = mrow >> 11;
            const int t  = mrow & (SEQ - 1);
            const int h  = col >> 6;
            const int dh = col & (D_HEAD - 1);
            const __bf16 v = (__bf16)accs[j][r];
            if (mat == 0)
                Q[(((size_t)(b * NUM_HEADS + h)) * SEQ + t) * D_HEAD + dh] = v;
            else if (mat == 1)
                K[(((size_t)(b * NUM_HEADS + h)) * SEQ + t) * D_HEAD + dh] = v;
            else
                Vt[(((size_t)(b * NUM_HEADS + h)) * D_HEAD + dh) * SEQ + t] = v;
        }
    }
}

// ---------------------------------------------------------------------------
// Kernel 2: causal flash attention with TDM double-buffered K/V staging.
// Block = (b, h, 64 query rows); 4 waves x 16 rows; keys in blocks of 32.
// Wave 0 drives the Tensor Data Mover; barriers publish LDS tiles.
// ---------------------------------------------------------------------------
__global__ __launch_bounds__(128) void attn_kernel(
    const __bf16* __restrict__ Q, const __bf16* __restrict__ K,
    const __bf16* __restrict__ Vt, __bf16* __restrict__ O) {
    __shared__ __bf16 kbuf[2][32 * 64];   // K tile: 32 keys x 64 d
    __shared__ __bf16 vbuf[2][64 * 32];   // V^T tile: 64 d x 32 keys
    __shared__ __bf16 pbuf[4][16][32];    // per-wave P re-layout staging

    const int w    = threadIdx.x >> 5;
    const int lane = threadIdx.x & 31;
    const int hw   = lane >> 4;
    const int ln   = lane & 15;

    const int nqb = SEQ / 64;
    const int bh  = blockIdx.x / nqb;
    const int qb  = blockIdx.x % nqb;
    const int q0  = qb * 64 + w * 16;
    const int b   = bh / NUM_HEADS;
    const int h   = bh % NUM_HEADS;

    const __bf16* Qh = Q  + (size_t)bh * SEQ * D_HEAD;
    const __bf16* Kh = K  + (size_t)bh * SEQ * D_HEAD;
    const __bf16* Vh = Vt + (size_t)bh * D_HEAD * SEQ;   // [Dh][S]

    const int nkb_blk = 2 * qb + 2;               // uniform over the block
    const int nkb_w   = (q0 + 15) / 32 + 1;       // this wave's causal bound

    // Kick off first K/V tiles via the Tensor Data Mover (wave 0 only).
    if (w == 0) {
        tdm_load_2d_b16((unsigned int)(uintptr_t)&kbuf[0][0], Kh,
                        D_HEAD, SEQ, D_HEAD, 32, D_HEAD);
        tdm_load_2d_b16((unsigned int)(uintptr_t)&vbuf[0][0], Vh,
                        SEQ, D_HEAD, 32, 64, SEQ);
    }

    const v16bf aq0 = load_frag_a_b16(Qh, D_HEAD, q0, 0);
    const v16bf aq1 = load_frag_a_b16(Qh, D_HEAD, q0, 32);

    v8f acc[4] = {v8f{}, v8f{}, v8f{}, v8f{}};
    float mrow[8], lrow[8];
#pragma unroll
    for (int r = 0; r < 8; ++r) { mrow[r] = -3.0e38f; lrow[r] = 0.0f; }

    const float scale = 0.125f;                   // 1/sqrt(64)
    int cur = 0;
    for (int kb = 0; kb < nkb_blk; ++kb) {
        if (w == 0) __builtin_amdgcn_s_wait_tensorcnt(0);
        __syncthreads();                          // tiles in buf[cur] ready

        if (w == 0 && kb + 1 < nkb_blk) {         // prefetch next tiles
            const int kn = (kb + 1) * 32;
            tdm_load_2d_b16((unsigned int)(uintptr_t)&kbuf[cur ^ 1][0],
                            Kh + (size_t)kn * D_HEAD,
                            D_HEAD, SEQ, D_HEAD, 32, D_HEAD);
            tdm_load_2d_b16((unsigned int)(uintptr_t)&vbuf[cur ^ 1][0],
                            Vh + kn, SEQ, D_HEAD, 32, 64, SEQ);
        }

        if (kb < nkb_w) {
            const int key0 = kb * 32;
            const __bf16* Kt = &kbuf[cur][0];     // ld = 64
            const __bf16* Vtl = &vbuf[cur][0];    // ld = 32

            v8f s0 = {}, s1 = {};
            s0 = wmma_bf16(aq0, load_frag_b_b16(Kt, 64,  0,  0), s0);
            s0 = wmma_bf16(aq1, load_frag_b_b16(Kt, 64,  0, 32), s0);
            s1 = wmma_bf16(aq0, load_frag_b_b16(Kt, 64, 16,  0), s1);
            s1 = wmma_bf16(aq1, load_frag_b_b16(Kt, 64, 16, 32), s1);

#pragma unroll
            for (int r = 0; r < 8; ++r) {
                const int qrow = q0 + r + 8 * hw;
                float v0 = s0[r] * scale;
                float v1 = s1[r] * scale;
                if (key0 + ln      > qrow) v0 = -3.0e38f;
                if (key0 + 16 + ln > qrow) v1 = -3.0e38f;

                float mx = fmaxf(v0, v1);
#pragma unroll
                for (int off = 1; off < 16; off <<= 1)
                    mx = fmaxf(mx, __shfl_xor(mx, off, 32));

                const float mnew = fmaxf(mrow[r], mx);
                const float corr = __expf(mrow[r] - mnew);
                const float e0 = __expf(v0 - mnew);
                const float e1 = __expf(v1 - mnew);
                float sum = e0 + e1;
#pragma unroll
                for (int off = 1; off < 16; off <<= 1)
                    sum += __shfl_xor(sum, off, 32);

                lrow[r] = lrow[r] * corr + sum;
                mrow[r] = mnew;
                acc[0][r] *= corr; acc[1][r] *= corr;
                acc[2][r] *= corr; acc[3][r] *= corr;

                pbuf[w][r + 8 * hw][ln]      = (__bf16)e0;
                pbuf[w][r + 8 * hw][16 + ln] = (__bf16)e1;
            }

            asm volatile("s_wait_dscnt 0" ::: "memory");  // wave-local P RAW

            const v16bf ap = load_frag_a_b16(&pbuf[w][0][0], 32, 0, 0);
#pragma unroll
            for (int dt = 0; dt < 4; ++dt) {
                const v16bf bv = load_frag_b_b16(Vtl, 32, dt * 16, 0);
                acc[dt] = wmma_bf16(ap, bv, acc[dt]);
            }
        }

        __syncthreads();                          // done reading buf[cur]
        cur ^= 1;
    }

#pragma unroll
    for (int r = 0; r < 8; ++r) {
        const float inv = 1.0f / lrow[r];
        const int t = q0 + r + 8 * hw;
        const size_t rowoff = ((size_t)b * SEQ + t) * D_MODEL + h * D_HEAD;
#pragma unroll
        for (int dt = 0; dt < 4; ++dt)
            O[rowoff + dt * 16 + ln] = (__bf16)(acc[dt][r] * inv);
    }
}

// ---------------------------------------------------------------------------
// Kernel 3: output projection y = O @ Wo^T (bf16 x bf16 -> f32), pipelined
// ---------------------------------------------------------------------------
__global__ __launch_bounds__(128) void out_proj_kernel(
    const __bf16* __restrict__ O, const __bf16* __restrict__ wo,
    float* __restrict__ y) {
    const int w   = threadIdx.x >> 5;
    const int wid = blockIdx.x * 4 + w;
    const int mt  = wid & 511;
    const int nt  = wid >> 9;
    const int n0  = nt * 64;

    const __bf16* arow = O + (size_t)(mt * 16 + (threadIdx.x & 15)) * D_MODEL;

    v8f acc0 = {}, acc1 = {}, acc2 = {}, acc3 = {};
    v16bf a_c  = load_frag_a_b16(O, D_MODEL, mt * 16, 0);
    v16bf b_c0 = load_frag_b_b16(wo, D_MODEL, n0 +  0, 0);
    v16bf b_c1 = load_frag_b_b16(wo, D_MODEL, n0 + 16, 0);
    v16bf b_c2 = load_frag_b_b16(wo, D_MODEL, n0 + 32, 0);
    v16bf b_c3 = load_frag_b_b16(wo, D_MODEL, n0 + 48, 0);

#pragma unroll 2
    for (int ks = 0; ks < D_MODEL / 32 - 1; ++ks) {
        const int kn = (ks + 1) * 32;
        const v16bf a_n  = load_frag_a_b16(O, D_MODEL, mt * 16, kn);
        const v16bf b_n0 = load_frag_b_b16(wo, D_MODEL, n0 +  0, kn);
        const v16bf b_n1 = load_frag_b_b16(wo, D_MODEL, n0 + 16, kn);
        const v16bf b_n2 = load_frag_b_b16(wo, D_MODEL, n0 + 32, kn);
        const v16bf b_n3 = load_frag_b_b16(wo, D_MODEL, n0 + 48, kn);
        __builtin_prefetch(arow + (ks + 2) * 32, 0, 3);
        acc0 = wmma_bf16(a_c, b_c0, acc0);
        acc1 = wmma_bf16(a_c, b_c1, acc1);
        acc2 = wmma_bf16(a_c, b_c2, acc2);
        acc3 = wmma_bf16(a_c, b_c3, acc3);
        a_c = a_n; b_c0 = b_n0; b_c1 = b_n1; b_c2 = b_n2; b_c3 = b_n3;
    }
    acc0 = wmma_bf16(a_c, b_c0, acc0);
    acc1 = wmma_bf16(a_c, b_c1, acc1);
    acc2 = wmma_bf16(a_c, b_c2, acc2);
    acc3 = wmma_bf16(a_c, b_c3, acc3);

    const int lane = threadIdx.x & 31;
    const int hw = lane >> 4, n = lane & 15;
    v8f accs[4] = {acc0, acc1, acc2, acc3};
#pragma unroll
    for (int j = 0; j < 4; ++j) {
#pragma unroll
        for (int r = 0; r < 8; ++r) {
            const int mrow = mt * 16 + r + 8 * hw;
            const int col  = n0 + j * 16 + n;
            y[(size_t)mrow * D_MODEL + col] = accs[j][r];
        }
    }
}

// ---------------------------------------------------------------------------
extern "C" void kernel_launch(void* const* d_in, const int* in_sizes, int n_in,
                              void* d_out, int out_size, void* d_ws,
                              size_t ws_size, hipStream_t stream) {
    (void)in_sizes; (void)n_in; (void)out_size; (void)ws_size;
    const float* x  = (const float*)d_in[0];
    const float* wq = (const float*)d_in[1];
    const float* wk = (const float*)d_in[2];
    const float* wv = (const float*)d_in[3];
    const float* wo = (const float*)d_in[4];
    float* y = (float*)d_out;

    const size_t elems = (size_t)BATCH * SEQ * D_MODEL;  // 8,388,608
    const size_t welem = (size_t)D_MODEL * D_MODEL;      // 1,048,576
    __bf16* p = (__bf16*)d_ws;
    __bf16* xb  = p; p += elems;
    __bf16* wqb = p; p += welem;
    __bf16* wkb = p; p += welem;
    __bf16* wvb = p; p += welem;
    __bf16* wob = p; p += welem;
    __bf16* Q   = p; p += elems;
    __bf16* K   = p; p += elems;
    __bf16* Vt  = p; p += elems;
    __bf16* O   = p;              // total ~92 MB of workspace

    cvt_kernel<<<(int)(elems / 4 / 256), 256, 0, stream>>>(x, xb, (int)(elems / 4));
    cvt_kernel<<<(int)(welem / 4 / 256), 256, 0, stream>>>(wq, wqb, (int)(welem / 4));
    cvt_kernel<<<(int)(welem / 4 / 256), 256, 0, stream>>>(wk, wkb, (int)(welem / 4));
    cvt_kernel<<<(int)(welem / 4 / 256), 256, 0, stream>>>(wv, wvb, (int)(welem / 4));
    cvt_kernel<<<(int)(welem / 4 / 256), 256, 0, stream>>>(wo, wob, (int)(welem / 4));

    qkv_proj_kernel<<<6144, 128, 0, stream>>>(xb, wqb, wkb, wvb, Q, K, Vt);
    attn_kernel<<<BATCH * NUM_HEADS * (SEQ / 64), 128, 0, stream>>>(Q, K, Vt, O);
    out_proj_kernel<<<2048, 128, 0, stream>>>(O, wob, y);
}
